// FreqTimeHPG_4174708211749
// MI455X (gfx1250) — matrix operand order repaired and checked by reading, not compiled, not run on hardware
//
#include <hip/hip_runtime.h>
#include <hip/hip_bf16.h>
#include <math.h>

typedef __attribute__((ext_vector_type(16))) _Float16 v16h;
typedef __attribute__((ext_vector_type(8)))  _Float16 v8h;
typedef __attribute__((ext_vector_type(8)))  float    v8f;

#define BATCH 16
#define SEQL  96
#define PREDL 96
#define NNODE 256
#define CFREQ 9
#define MNODE (NNODE * CFREQ)        // 2304
#define BM    (BATCH * MNODE)        // 36864
#define EMB   128
#define HID   64
#define KNN   8
#define BN    (BATCH * NNODE)        // 4096
#define PI_F  3.14159265358979323846f

__device__ __forceinline__ float silu_f(float x) { return x / (1.f + __expf(-x)); }

// merge two contiguous v8h (b128) loads into one WMMA fragment
__device__ __forceinline__ v16h frag16(const _Float16* p0, const _Float16* p1) {
  v8h a0 = *(const v8h*)p0;
  v8h a1 = *(const v8h*)p1;
  return __builtin_shufflevector(a0, a1, 0, 1, 2, 3, 4, 5, 6, 7, 8, 9, 10, 11, 12, 13, 14, 15);
}

// ---------------- trend = moving_avg(x), MA_K=3 with edge replication ---------
__global__ void moving_avg_k(const float* __restrict__ x, float* __restrict__ trend, int total) {
  int t = blockIdx.x * blockDim.x + threadIdx.x;
  if (t >= total) return;
  int n  = t % NNODE;
  int tt = (t / NNODE) % SEQL;
  int b  = t / (NNODE * SEQL);
  int tm = tt > 0 ? tt - 1 : 0;
  int tp = tt < SEQL - 1 ? tt + 1 : SEQL - 1;
  size_t base = (size_t)b * SEQL * NNODE;
  trend[t] = (x[base + (size_t)tm * NNODE + n] + x[t] + x[base + (size_t)tp * NNODE + n]) * (1.f / 3.f);
}

// ---------------- 16-pt rfft (ortho) of seasonal[:, :16, :], out (b, n, c) ----
__global__ void rfft16_k(const float* __restrict__ x, const float* __restrict__ trend,
                         float* __restrict__ Sre, float* __restrict__ Sim, int total) {
  int t = blockIdx.x * blockDim.x + threadIdx.x;
  if (t >= total) return;                     // t = (b*NNODE + n)*CFREQ + k
  int k = t % CFREQ;
  int n = (t / CFREQ) % NNODE;
  int b = t / (CFREQ * NNODE);
  float sr = 0.f, si = 0.f;
  for (int tau = 0; tau < 16; ++tau) {
    size_t xo = ((size_t)b * SEQL + tau) * NNODE + n;
    float s = x[xo] - trend[xo];
    float ang = -2.f * PI_F * (float)(k * tau) / 16.f;
    sr += s * __cosf(ang);
    si += s * __sinf(ang);
  }
  Sre[t] = sr * 0.25f;  // 1/sqrt(16)
  Sim[t] = si * 0.25f;
}

// ---------------- per-sample 8-NN in (re,im) plane ----------------------------
__global__ void knn_k(const float* __restrict__ Sre, const float* __restrict__ Sim,
                      int* __restrict__ knn) {
  __shared__ float fr[256], fi[256];
  int b = blockIdx.x / (MNODE / 256);
  int i = (blockIdx.x % (MNODE / 256)) * 256 + threadIdx.x;
  int m = b * MNODE + i;
  float xr = Sre[m], xi = Sim[m];
  float sqi = xr * xr + xi * xi;
  float bd[KNN]; int bi[KNN];
  #pragma unroll
  for (int k = 0; k < KNN; ++k) { bd[k] = INFINITY; bi[k] = 0; }
  for (int jt = 0; jt < MNODE / 256; ++jt) {
    __syncthreads();
    int j0 = jt * 256;
    fr[threadIdx.x] = Sre[b * MNODE + j0 + threadIdx.x];
    fi[threadIdx.x] = Sim[b * MNODE + j0 + threadIdx.x];
    __syncthreads();
    for (int jj = 0; jj < 256; ++jj) {
      int j = j0 + jj;
      float yr = fr[jj], yi = fi[jj];
      float d = sqi + yr * yr + yi * yi - 2.f * (xr * yr + xi * yi);
      if (j == i) d = INFINITY;
      if (d < bd[KNN - 1]) {
        int k = KNN - 2;
        while (k >= 0 && d < bd[k]) { bd[k + 1] = bd[k]; bi[k + 1] = bi[k]; --k; }
        bd[k + 1] = d; bi[k + 1] = j;
      }
    }
  }
  #pragma unroll
  for (int k = 0; k < KNN; ++k) knn[(size_t)m * KNN + k] = bi[k];
}

__global__ void zero_f32_k(float* __restrict__ p, size_t n) {
  size_t t = (size_t)blockIdx.x * blockDim.x + threadIdx.x;
  if (t < n) p[t] = 0.f;
}

__global__ void deg_count_k(const int* __restrict__ knn, float* __restrict__ deg, int total) {
  int t = blockIdx.x * blockDim.x + threadIdx.x;   // t over BM*KNN
  if (t >= total) return;
  int ig = t / KNN;
  int b  = ig / MNODE;
  int mj = b * MNODE + knn[t];
  atomicAdd(&deg[ig], 1.f);
  atomicAdd(&deg[mj], 1.f);
}

// ---------------- node features: X0 = S ⊗ freq_emb ---------------------------
__global__ void build_x0_k(const float* __restrict__ Sre, const float* __restrict__ Sim,
                           const float* __restrict__ femb,
                           float* __restrict__ X0r, float* __restrict__ X0i, size_t total) {
  size_t t = (size_t)blockIdx.x * blockDim.x + threadIdx.x;
  if (t >= total) return;
  int e = (int)(t & (EMB - 1));
  size_t m = t >> 7;
  int c = (int)(m % CFREQ);
  float f = femb[c * EMB + e];
  X0r[t] = Sre[m] * f;
  X0i[t] = Sim[m] * f;
}

// ---------------- out[r] += w * in[c] over symmetrised edges ------------------
__global__ void scatter_prop_k(const float* __restrict__ inR, const float* __restrict__ inI,
                               float* __restrict__ outR, float* __restrict__ outI,
                               const int* __restrict__ knn, const float* __restrict__ deg,
                               size_t total) {
  size_t t = (size_t)blockIdx.x * blockDim.x + threadIdx.x;
  if (t >= total) return;
  int e = (int)(t & (EMB - 1));
  size_t edge = t >> 7;               // edge over BM*KNN
  size_t mi = edge / KNN;
  int b = (int)(mi / MNODE);
  size_t mj = (size_t)b * MNODE + knn[edge];
  float w = rsqrtf(deg[mi] + 1e-8f) * rsqrtf(deg[mj] + 1e-8f) * 0.5f;  // /S_SCALE
  atomicAdd(&outR[mi * EMB + e], w * inR[mj * EMB + e]);
  atomicAdd(&outI[mi * EMB + e], w * inI[mj * EMB + e]);
  atomicAdd(&outR[mj * EMB + e], w * inR[mi * EMB + e]);
  atomicAdd(&outI[mj * EMB + e], w * inI[mi * EMB + e]);
}

// acc = (c0-c2)*T0 - c1*P1 + 2*c2*P2 (written into X0)
__global__ void cheb_combine_k(float* __restrict__ X0, const float* __restrict__ P1,
                               const float* __restrict__ P2, const float* __restrict__ approx,
                               const float* __restrict__ theta, size_t n) {
  size_t t = (size_t)blockIdx.x * blockDim.x + threadIdx.x;
  if (t >= n) return;
  float c0 = theta[0] * approx[0] + theta[1] * approx[3];
  float c1 = theta[0] * approx[1] + theta[1] * approx[4];
  float c2 = theta[0] * approx[2] + theta[1] * approx[5];
  X0[t] = (c0 - c2) * X0[t] - c1 * P1[t] + 2.f * c2 * P2[t];
}

__global__ void cvt_f16_k(const float* __restrict__ in, _Float16* __restrict__ out, size_t n) {
  size_t t = (size_t)blockIdx.x * blockDim.x + threadIdx.x;
  if (t < n) out[t] = (_Float16)in[t];
}

// Pack W (OutN x K row-major, f32) into WMMA-B fragment order, with scale:
// Bp[((kt*tilesN + nt)*32 + lane)*16 + h] = scale * W[n*K + k]
//   n = nt*16 + (lane&15), k = kt*32 + (lane>=16 ? 16 : 0) + h
__global__ void packB_k(const float* __restrict__ W, _Float16* __restrict__ Bp,
                        int OutN, int K, float scale) {
  int t = blockIdx.x * blockDim.x + threadIdx.x;
  if (t >= OutN * K) return;
  int h    = t & 15;
  int lane = (t >> 4) & 31;
  int tile = t >> 9;
  int tilesN = OutN >> 4;
  int kt = tile / tilesN;
  int nt = tile % tilesN;
  int n = (nt << 4) + (lane & 15);
  int k = (kt << 5) + ((lane >> 4) << 4) + h;
  Bp[t] = (_Float16)(scale * W[(size_t)n * K + k]);
}

__global__ void trend_T_f16_k(const float* __restrict__ trend, _Float16* __restrict__ out, int total) {
  int t = blockIdx.x * blockDim.x + threadIdx.x;   // t = (b*NNODE+n)*SEQL+tt
  if (t >= total) return;
  int tt = t % SEQL;
  int n  = (t / SEQL) % NNODE;
  int b  = t / (SEQL * NNODE);
  out[t] = (_Float16)trend[((size_t)b * SEQL + tt) * NNODE + n];
}

// ======= fused complex GEMM: Hr + iHi = (Ar + iAi) @ (Wr + iWi)^T, K=N=128 ===
// Weights pre-packed in fragment order and staged in LDS; 2 row-tiles/wave.
__global__ void cgemm_wmma_k(const _Float16* __restrict__ Ar, const _Float16* __restrict__ Ai,
                             const _Float16* __restrict__ BpR, const _Float16* __restrict__ BpI,
                             float* __restrict__ Hr, float* __restrict__ Hi, int M) {
  __shared__ __align__(16) _Float16 sWr[EMB * EMB];
  __shared__ __align__(16) _Float16 sWi[EMB * EMB];
  for (int idx = threadIdx.x * 8; idx < EMB * EMB; idx += blockDim.x * 8) {
    *(v8h*)(sWr + idx) = *(const v8h*)(BpR + idx);
    *(v8h*)(sWi + idx) = *(const v8h*)(BpI + idx);
  }
  __syncthreads();
  const int tilesN = EMB / 16;  // 8
  int wave = blockIdx.x * (blockDim.x >> 5) + (threadIdx.x >> 5);
  int lane = threadIdx.x & 31;
  int total = (M >> 5) * tilesN;
  if (wave >= total) return;                 // wave-uniform
  int nt = wave % tilesN;
  int tm = (wave / tilesN) << 5;             // 2 row-tiles
  int ncol = (nt << 4) + (lane & 15);
  int kA = (lane >> 4) << 4;                 // 0 / 16
  v8f accR0 = {}, accR1 = {}, accI0 = {}, accI1 = {};
  #pragma unroll
  for (int kt = 0; kt < EMB / 32; ++kt) {
    int boff = ((kt * tilesN + nt) * 32 + lane) * 16;
    v16h br = frag16(sWr + boff, sWr + boff + 8);
    v16h bi = frag16(sWi + boff, sWi + boff + 8);
    v16h bin = -bi;                          // f16 WMMA has no A/B NEG: negate operand
    size_t arow0 = (size_t)(tm + (lane & 15)) * EMB + kt * 32 + kA;
    size_t arow1 = arow0 + (size_t)16 * EMB;
    v16h ar0 = frag16(Ar + arow0, Ar + arow0 + 8);
    v16h ai0 = frag16(Ai + arow0, Ai + arow0 + 8);
    v16h ar1 = frag16(Ar + arow1, Ar + arow1 + 8);
    v16h ai1 = frag16(Ai + arow1, Ai + arow1 + 8);
    accR0 = __builtin_amdgcn_wmma_f32_16x16x32_f16(false, ar0, false, br,  (short)0, accR0, false, false);
    accR0 = __builtin_amdgcn_wmma_f32_16x16x32_f16(false, ai0, false, bin, (short)0, accR0, false, false);
    accI0 = __builtin_amdgcn_wmma_f32_16x16x32_f16(false, ar0, false, bi,  (short)0, accI0, false, false);
    accI0 = __builtin_amdgcn_wmma_f32_16x16x32_f16(false, ai0, false, br,  (short)0, accI0, false, false);
    accR1 = __builtin_amdgcn_wmma_f32_16x16x32_f16(false, ar1, false, br,  (short)0, accR1, false, false);
    accR1 = __builtin_amdgcn_wmma_f32_16x16x32_f16(false, ai1, false, bin, (short)0, accR1, false, false);
    accI1 = __builtin_amdgcn_wmma_f32_16x16x32_f16(false, ar1, false, bi,  (short)0, accI1, false, false);
    accI1 = __builtin_amdgcn_wmma_f32_16x16x32_f16(false, ai1, false, br,  (short)0, accI1, false, false);
  }
  int rb0 = tm + ((lane >> 4) << 3);
  #pragma unroll
  for (int v = 0; v < 8; ++v) {
    size_t o0 = (size_t)(rb0 + v) * EMB + ncol;
    size_t o1 = (size_t)(rb0 + 16 + v) * EMB + ncol;
    Hr[o0] = accR0[v];
    Hi[o0] = accI0[v];
    Hr[o1] = accR1[v];
    Hi[o1] = accI1[v];
  }
}

// ======= generic WMMA GEMM: C = beta*C + alpha*A@B + bias; B packed, in LDS ==
// 4 row-tiles per wave (M % 64 == 0), K is compile-time (full unroll).
template <int K>
__global__ void gemm_wmma_k(const _Float16* __restrict__ A, const _Float16* __restrict__ Bp,
                            float* __restrict__ C, int M, int N,
                            float alpha, float beta, const float* __restrict__ bias) {
  extern __shared__ _Float16 sB[];
  int nhalf = K * N;
  for (int idx = threadIdx.x * 8; idx < nhalf; idx += blockDim.x * 8)
    *(v8h*)(sB + idx) = *(const v8h*)(Bp + idx);
  __syncthreads();
  int wave = blockIdx.x * (blockDim.x >> 5) + (threadIdx.x >> 5);
  int lane = threadIdx.x & 31;
  int tilesN = N >> 4;
  int total = (M >> 6) * tilesN;
  if (wave >= total) return;                 // wave-uniform
  int nt = wave % tilesN;
  int tm = (wave / tilesN) << 6;             // 4 row-tiles
  int ncol = (nt << 4) + (lane & 15);
  int kA = (lane >> 4) << 4;
  v8f acc[4] = {};
  #pragma unroll
  for (int kt = 0; kt < K / 32; ++kt) {
    int boff = ((kt * tilesN + nt) * 32 + lane) * 16;
    v16h b = frag16(sB + boff, sB + boff + 8);
    #pragma unroll
    for (int r = 0; r < 4; ++r) {
      const _Float16* ap = A + (size_t)(tm + r * 16 + (lane & 15)) * K + kt * 32 + kA;
      v16h a = frag16(ap, ap + 8);
      acc[r] = __builtin_amdgcn_wmma_f32_16x16x32_f16(false, a, false, b, (short)0, acc[r], false, false);
    }
  }
  #pragma unroll
  for (int r = 0; r < 4; ++r) {
    int rbase = tm + r * 16 + ((lane >> 4) << 3);
    #pragma unroll
    for (int v = 0; v < 8; ++v) {
      size_t off = (size_t)(rbase + v) * N + ncol;
      float val = alpha * acc[r][v];
      if (bias) val += bias[ncol];
      if (beta != 0.f) val += beta * C[off];
      C[off] = val;
    }
  }
}

// -------- complex SiLU then E->1 complex projection (one wave per row) -------
__global__ void silu_eo_k(const float* __restrict__ hr, const float* __restrict__ hi,
                          const float* __restrict__ eoWr, const float* __restrict__ eoWi,
                          float* __restrict__ Sre, float* __restrict__ Sim, int rows) {
  int wave = blockIdx.x * (blockDim.x >> 5) + (threadIdx.x >> 5);
  int lane = threadIdx.x & 31;
  if (wave >= rows) return;
  float ar = 0.f, ai = 0.f;
  for (int e = lane; e < EMB; e += 32) {
    float xr = silu_f(hr[(size_t)wave * EMB + e]);
    float xi = silu_f(hi[(size_t)wave * EMB + e]);
    float wr = eoWr[e], wi = eoWi[e];
    ar += xr * wr - xi * wi;
    ai += xr * wi + xi * wr;
  }
  #pragma unroll
  for (int off = 16; off > 0; off >>= 1) {
    ar += __shfl_xor(ar, off, 32);
    ai += __shfl_xor(ai, off, 32);
  }
  if (lane == 0) { Sre[wave] = ar; Sim[wave] = ai; }
}

// -------- 96-pt irfft (ortho) from 9 bins, H (b,n,t) -------------------------
__global__ void irfft96_k(const float* __restrict__ Sre, const float* __restrict__ Sim,
                          float* __restrict__ H, int total) {
  int t = blockIdx.x * blockDim.x + threadIdx.x;   // t = (b*NNODE+n)*SEQL+tt
  if (t >= total) return;
  int tt = t % SEQL;
  int n  = (t / SEQL) % NNODE;
  int b  = t / (SEQL * NNODE);
  size_t base = (size_t)b * MNODE + (size_t)n * CFREQ;
  float acc = Sre[base];
  #pragma unroll
  for (int k = 1; k < CFREQ; ++k) {
    float ang = 2.f * PI_F * (float)(k * tt) / (float)SEQL;
    acc += 2.f * (Sre[base + k] * __cosf(ang) - Sim[base + k] * __sinf(ang));
  }
  H[t] = acc * 0.1020620726f;  // 1/sqrt(96)
}

// -------- instance norm over L (+ per-node affine) then SiLU, in place -------
__global__ void inorm_silu_k(float* __restrict__ Hbuf, int L,
                             const float* __restrict__ w, const float* __restrict__ bb,
                             int rows) {
  int wave = blockIdx.x * (blockDim.x >> 5) + (threadIdx.x >> 5);
  int lane = threadIdx.x & 31;
  if (wave >= rows) return;
  float* row = Hbuf + (size_t)wave * L;
  float s = 0.f, s2 = 0.f;
  for (int t = lane; t < L; t += 32) { float v = row[t]; s += v; s2 += v * v; }
  #pragma unroll
  for (int off = 16; off > 0; off >>= 1) {
    s  += __shfl_xor(s,  off, 32);
    s2 += __shfl_xor(s2, off, 32);
  }
  float mu  = s / (float)L;
  float var = s2 / (float)L - mu * mu;
  float inv = rsqrtf(var + 1e-5f);
  int n = wave % NNODE;
  float ww = w[n], b0 = bb[n];
  for (int t = lane; t < L; t += 32) {
    float v = (row[t] - mu) * inv * ww + b0;
    row[t] = silu_f(v);
  }
}

// ============================ host side ======================================
static inline void* wsalloc(char*& p, size_t bytes) {
  void* r = (void*)p;
  p += (bytes + 255) & ~(size_t)255;
  return r;
}
static inline unsigned gdim(size_t n, int bs) { return (unsigned)((n + bs - 1) / bs); }

extern "C" void kernel_launch(void* const* d_in, const int* in_sizes, int n_in,
                              void* d_out, int out_size, void* d_ws, size_t ws_size,
                              hipStream_t stream) {
  (void)in_sizes; (void)n_in; (void)out_size; (void)ws_size;
  const float* x      = (const float*)d_in[0];
  const float* approx = (const float*)d_in[1];   // [2,3]
  const float* theta  = (const float*)d_in[2];   // [2]
  const float* fr_Wr  = (const float*)d_in[3];   // [128,128]
  const float* fr_Wi  = (const float*)d_in[4];
  const float* eo_Wr  = (const float*)d_in[5];   // [1,128]
  const float* eo_Wi  = (const float*)d_in[6];
  const float* femb   = (const float*)d_in[7];   // [9,128]
  const float* n1_w = (const float*)d_in[8];  const float* n1_b = (const float*)d_in[9];
  const float* n2_w = (const float*)d_in[10]; const float* n2_b = (const float*)d_in[11];
  const float* W1_w = (const float*)d_in[12]; const float* W1_b = (const float*)d_in[13];
  const float* W2_w = (const float*)d_in[14]; const float* W2_b = (const float*)d_in[15];
  const float* Wt_w = (const float*)d_in[16]; const float* Wt_b = (const float*)d_in[17];
  const float* W3_w = (const float*)d_in[18]; const float* W3_b = (const float*)d_in[19];
  float* out = (float*)d_out;

  char* p = (char*)d_ws;
  float* trend = (float*)wsalloc(p, (size_t)BATCH * SEQL * NNODE * 4);
  float* Sre   = (float*)wsalloc(p, (size_t)BM * 4);
  float* Sim   = (float*)wsalloc(p, (size_t)BM * 4);
  int*   knn   = (int*)  wsalloc(p, (size_t)BM * KNN * 4);
  float* deg   = (float*)wsalloc(p, (size_t)BM * 4);
  float* X0r = (float*)wsalloc(p, (size_t)BM * EMB * 4);
  float* X0i = (float*)wsalloc(p, (size_t)BM * EMB * 4);
  float* X1r = (float*)wsalloc(p, (size_t)BM * EMB * 4);
  float* X1i = (float*)wsalloc(p, (size_t)BM * EMB * 4);
  float* X2r = (float*)wsalloc(p, (size_t)BM * EMB * 4);
  float* X2i = (float*)wsalloc(p, (size_t)BM * EMB * 4);
  _Float16* WrT = (_Float16*)wsalloc(p, (size_t)EMB * EMB * 2);
  _Float16* WiT = (_Float16*)wsalloc(p, (size_t)EMB * EMB * 2);
  _Float16* W1T = (_Float16*)wsalloc(p, (size_t)SEQL * HID * 2);
  _Float16* W2T = (_Float16*)wsalloc(p, (size_t)HID * HID * 2);
  _Float16* WtT = (_Float16*)wsalloc(p, (size_t)SEQL * HID * 2);
  _Float16* W3T = (_Float16*)wsalloc(p, (size_t)HID * PREDL * 2);
  float*    H1  = (float*)   wsalloc(p, (size_t)BN * SEQL * 4);
  _Float16* Hh  = (_Float16*)wsalloc(p, (size_t)BN * SEQL * 2);
  float*    H2  = (float*)   wsalloc(p, (size_t)BN * HID * 4);
  _Float16* H2h = (_Float16*)wsalloc(p, (size_t)BN * HID * 2);
  float*    H3  = (float*)   wsalloc(p, (size_t)BN * HID * 4);
  _Float16* H3h = (_Float16*)wsalloc(p, (size_t)BN * HID * 2);
  _Float16* tTh = (_Float16*)wsalloc(p, (size_t)BN * SEQL * 2);
  // buffer reuse after the Chebyshev combine frees X1/X2:
  _Float16* aHr = (_Float16*)X1r;
  _Float16* aHi = (_Float16*)X1i;
  float* hr = X2r;
  float* hi = X2i;
  float* SoRe = Sre;   // S no longer needed after graph + features built
  float* SoIm = Sim;

  const int BS = 256;

  // 1. decomposition + spectrum
  moving_avg_k<<<gdim((size_t)BATCH * SEQL * NNODE, BS), BS, 0, stream>>>(x, trend, BATCH * SEQL * NNODE);
  rfft16_k<<<gdim((size_t)BM, BS), BS, 0, stream>>>(x, trend, Sre, Sim, BM);

  // 2. graph build
  knn_k<<<BATCH * (MNODE / 256), 256, 0, stream>>>(Sre, Sim, knn);
  zero_f32_k<<<gdim((size_t)BM, BS), BS, 0, stream>>>(deg, (size_t)BM);
  deg_count_k<<<gdim((size_t)BM * KNN, BS), BS, 0, stream>>>(knn, deg, BM * KNN);

  // 3. node features + Chebyshev propagation (P1 = A(T0), P2 = A(P1))
  size_t nX = (size_t)BM * EMB;
  build_x0_k<<<gdim(nX, BS), BS, 0, stream>>>(Sre, Sim, femb, X0r, X0i, nX);
  zero_f32_k<<<gdim(nX, BS), BS, 0, stream>>>(X1r, nX);
  zero_f32_k<<<gdim(nX, BS), BS, 0, stream>>>(X1i, nX);
  size_t nEdgeE = (size_t)BM * KNN * EMB;
  scatter_prop_k<<<gdim(nEdgeE, BS), BS, 0, stream>>>(X0r, X0i, X1r, X1i, knn, deg, nEdgeE);
  zero_f32_k<<<gdim(nX, BS), BS, 0, stream>>>(X2r, nX);
  zero_f32_k<<<gdim(nX, BS), BS, 0, stream>>>(X2i, nX);
  scatter_prop_k<<<gdim(nEdgeE, BS), BS, 0, stream>>>(X1r, X1i, X2r, X2i, knn, deg, nEdgeE);
  cheb_combine_k<<<gdim(nX, BS), BS, 0, stream>>>(X0r, X1r, X2r, approx, theta, nX);
  cheb_combine_k<<<gdim(nX, BS), BS, 0, stream>>>(X0i, X1i, X2i, approx, theta, nX);

  // 4. fused complex linear (one kernel, 8 WMMA per k-step)
  cvt_f16_k<<<gdim(nX, BS), BS, 0, stream>>>(X0r, aHr, nX);
  cvt_f16_k<<<gdim(nX, BS), BS, 0, stream>>>(X0i, aHi, nX);
  packB_k<<<gdim((size_t)EMB * EMB, BS), BS, 0, stream>>>(fr_Wr, WrT, EMB, EMB, 1.f);
  packB_k<<<gdim((size_t)EMB * EMB, BS), BS, 0, stream>>>(fr_Wi, WiT, EMB, EMB, 1.f);
  {
    int totalWaves = (BM / 32) * (EMB / 16);           // 9216
    cgemm_wmma_k<<<gdim((size_t)totalWaves, 8), 256, 0, stream>>>(aHr, aHi, WrT, WiT, hr, hi, BM);
  }

  // 5. complex SiLU + E->1 projection, then 96-pt irfft
  silu_eo_k<<<gdim((size_t)BM, 8), 256, 0, stream>>>(hr, hi, eo_Wr, eo_Wi, SoRe, SoIm, BM);
  irfft96_k<<<gdim((size_t)BN * SEQL, BS), BS, 0, stream>>>(SoRe, SoIm, H1, BN * SEQL);

  // 6. time-domain MLP chain (WMMA GEMMs, weights staged in LDS)
  inorm_silu_k<<<gdim((size_t)BN, 8), 256, 0, stream>>>(H1, SEQL, n1_w, n1_b, BN);
  cvt_f16_k<<<gdim((size_t)BN * SEQL, BS), BS, 0, stream>>>(H1, Hh, (size_t)BN * SEQL);
  packB_k<<<gdim((size_t)HID * SEQL, BS), BS, 0, stream>>>(W1_w, W1T, HID, SEQL, 1.f);
  gemm_wmma_k<SEQL><<<gdim((size_t)(BN / 64) * (HID / 16), 8), 256, SEQL * HID * 2, stream>>>(
      Hh, W1T, H2, BN, HID, 1.f, 0.f, W1_b);

  inorm_silu_k<<<gdim((size_t)BN, 8), 256, 0, stream>>>(H2, HID, n2_w, n2_b, BN);
  cvt_f16_k<<<gdim((size_t)BN * HID, BS), BS, 0, stream>>>(H2, H2h, (size_t)BN * HID);
  packB_k<<<gdim((size_t)HID * HID, BS), BS, 0, stream>>>(W2_w, W2T, HID, HID, 1.f);
  gemm_wmma_k<HID><<<gdim((size_t)(BN / 64) * (HID / 16), 8), 256, HID * HID * 2, stream>>>(
      H2h, W2T, H3, BN, HID, 1.f, 0.f, W2_b);

  trend_T_f16_k<<<gdim((size_t)BN * SEQL, BS), BS, 0, stream>>>(trend, tTh, BN * SEQL);
  packB_k<<<gdim((size_t)HID * SEQL, BS), BS, 0, stream>>>(Wt_w, WtT, HID, SEQL, 1.f);
  gemm_wmma_k<SEQL><<<gdim((size_t)(BN / 64) * (HID / 16), 8), 256, SEQL * HID * 2, stream>>>(
      tTh, WtT, H3, BN, HID, 1.f, 1.f, Wt_b);           // H3 += trend_emb + Wt_b

  cvt_f16_k<<<gdim((size_t)BN * HID, BS), BS, 0, stream>>>(H3, H3h, (size_t)BN * HID);
  packB_k<<<gdim((size_t)PREDL * HID, BS), BS, 0, stream>>>(W3_w, W3T, PREDL, HID, 1.f);
  gemm_wmma_k<HID><<<gdim((size_t)(BN / 64) * (PREDL / 16), 8), 256, HID * PREDL * 2, stream>>>(
      H3h, W3T, out, BN, PREDL, 1.f, 0.f, W3_b);
}
